// AutoEncoderBatch_1614907703390
// MI455X (gfx1250) — matrix operand order repaired and checked by reading, not compiled, not run on hardware
//
#include <hip/hip_runtime.h>

// Persistent-kernel GRU seq2seq autoencoder for gfx1250 (MI455X).
// B=256, T=1024, H=512, IN=1. bf16 WMMA (16x16x32) with f32 accumulation.
// fp32 hidden state lives in registers (same thread owns (row,j) every step);
// only the bf16 WMMA A-operand copy of h is shared through global/L2.

namespace {
constexpr int Bn  = 256;
constexpr int Tn  = 1024;
constexpr int Hn  = 512;
constexpr int NWG = 32;            // one WG per 16-wide hidden block (H/16 = 32)
constexpr int BLK = 256;           // 8 wave32 waves
constexpr int GSZ = NWG * BLK;
}

typedef __attribute__((ext_vector_type(16))) __bf16 v16bf;
typedef __attribute__((ext_vector_type(8)))  float  v8f;

union Frag {
  uint4    q[2];
  unsigned u[8];
  v16bf    v;
};

__device__ __forceinline__ unsigned f2bf(float f) {
  unsigned u = __float_as_uint(f);
  return (u + 0x7FFFu + ((u >> 16) & 1u)) >> 16;   // RNE fp32 -> bf16
}
// v_exp_f32 + v_rcp_f32 fast activations (no IEEE div expansion)
__device__ __forceinline__ float sigm_fast(float v) {
  return __builtin_amdgcn_rcpf(1.0f + __expf(-v));
}
__device__ __forceinline__ float tanh_fast(float v) {
  return 2.0f * __builtin_amdgcn_rcpf(1.0f + __expf(-2.0f * v)) - 1.0f;
}

__global__ __launch_bounds__(BLK, 1) void gru_autoencoder_kernel(
    const float* __restrict__ input, const int* __restrict__ seqlen,
    const float* __restrict__ eWih, const float* __restrict__ eWhh,
    const float* __restrict__ ebih, const float* __restrict__ ebhh,
    const float* __restrict__ dWih, const float* __restrict__ dWhh,
    const float* __restrict__ dbih, const float* __restrict__ dbhh,
    const float* __restrict__ Wout, const float* __restrict__ bout,
    float* __restrict__ out_loss, float* __restrict__ out_feat,
    float* __restrict__ out_in,   float* __restrict__ out_y,
    unsigned short* hb0, unsigned short* hb1,
    float* xpart, float* lpart, unsigned* syncc)
{
  const int wg   = blockIdx.x;
  const int tid  = threadIdx.x;
  const int lane = tid & 31;
  const int wave = tid >> 5;
  const int j0   = wg << 4;            // this WG's hidden-unit block [j0, j0+16)
  const int gtid = wg * BLK + tid;

  // B-matrix fragments for 3 gates, pre-swizzled to WMMA per-lane layout:
  // [gate][kk][v][lane] packed bf16 pairs -> 3*16*8*32 dwords = 48 KB
  __shared__ unsigned lds_w[3 * 16 * 8 * 32];
  __shared__ float x_lds[Bn];
  __shared__ float wih_l[48];          // Wih columns, 3 gates x 16
  __shared__ float brz_l[32];          // bih+bhh folded, r/z gates
  __shared__ float bin_l[16];          // bih, n gate
  __shared__ float bhn_l[16];          // bhh, n gate
  __shared__ float wout_l[16];
  __shared__ float red1[BLK], red2[BLK];

  unsigned sync_no = 0;
  auto gsync = [&]() {
    __threadfence();
    __syncthreads();
    ++sync_no;
    if (tid == 0) {
      atomicAdd(syncc, 1u);
      const unsigned goal = sync_no * (unsigned)NWG;
      while (__hip_atomic_load(syncc, __ATOMIC_RELAXED, __HIP_MEMORY_SCOPE_AGENT) < goal)
        __builtin_amdgcn_s_sleep(1);
    }
    __syncthreads();
    __threadfence();
  };

  // Stage Whh^T slice for this WG's 3 gate columns into LDS as WMMA-B fragments.
  // B is K x N (K = hidden in, N = 16 units of one gate). B k order used here (same as A):
  // lanes 0-15 hold K=0..15 (VGPR v -> K=2v,2v+1), lanes 16-31 hold K=16..31.
  auto stage = [&](const float* Whh, const float* Wih, const float* bih, const float* bhh) {
    for (int i = tid; i < 3 * 16 * 32 * 8; i += BLK) {
      const int v  = i & 7;
      const int ln = (i >> 3) & 31;
      const int r  = i >> 8;
      const int kk = r & 15;
      const int g  = r >> 4;
      const int n  = ln & 15;
      const int k  = kk * 32 + ((ln >> 4) << 4) + 2 * v;
      const float* wr = Whh + (size_t)(g * Hn + j0 + n) * Hn;
      lds_w[(((g * 16 + kk) * 8 + v) << 5) + ln] = f2bf(wr[k]) | (f2bf(wr[k + 1]) << 16);
    }
    if (tid < 48) {
      const int g = tid >> 4, c = tid & 15;
      const int idx = g * Hn + j0 + c;     // Wih is [3H, 1]
      wih_l[tid] = Wih[idx];
      if (g < 2) brz_l[tid] = bih[idx] + bhh[idx];   // fold biases for r, z
      else { bin_l[c] = bih[idx]; bhn_l[c] = bhh[idx]; }
    }
  };

  // fp32 hidden state in registers: hreg[mt*8+e] <-> (row = mt-tile row, col = j0+lane%16)
  float hreg[16];
#pragma unroll
  for (int i = 0; i < 16; ++i) hreg[i] = 0.0f;
  int slreg[16];
  {
    const int hi = lane >> 4;
#pragma unroll
    for (int mt = 0; mt < 2; ++mt)
#pragma unroll
      for (int e = 0; e < 8; ++e)
        slreg[mt * 8 + e] = seqlen[((wave * 2 + mt) << 4) + e + (hi << 3)];
  }

  // One GRU time step: gh = h @ Whh^T (bf16 WMMA), fuse gates in registers.
  auto do_step = [&](const unsigned short* hcb, unsigned short* hnb,
                     float* xp_out, bool enc, int t) {
    const int c  = lane & 15;
    const int j  = j0 + c;
    const int hi = lane >> 4;
#pragma unroll
    for (int mt = 0; mt < 2; ++mt) {
      const int m0   = (wave * 2 + mt) << 4;   // 8 waves x 2 tiles cover all 256 rows
      const int arow = m0 + c;
      v8f aR = {}, aZ = {}, aN = {};
      for (int kk = 0; kk < 16; ++kk) {
        // A fragment (16-bit 16x32): lanes 0-15 K=0..15, lanes 16-31 K=16..31.
        Frag fa;
        const unsigned short* pa = hcb + (size_t)arow * Hn + kk * 32 + (hi << 4);
        fa.q[0] = *(const uint4*)pa;
        fa.q[1] = *(const uint4*)(pa + 8);
        Frag fr, fz, fn;
        const int b0 = ((kk * 8) << 5) + lane;
        const int b1 = (((16 + kk) * 8) << 5) + lane;
        const int b2 = (((32 + kk) * 8) << 5) + lane;
#pragma unroll
        for (int v = 0; v < 8; ++v) {
          fr.u[v] = lds_w[b0 + (v << 5)];
          fz.u[v] = lds_w[b1 + (v << 5)];
          fn.u[v] = lds_w[b2 + (v << 5)];
        }
        aR = __builtin_amdgcn_wmma_f32_16x16x32_bf16(false, fa.v, false, fr.v, (short)0, aR, false, false);
        aZ = __builtin_amdgcn_wmma_f32_16x16x32_bf16(false, fa.v, false, fz.v, (short)0, aZ, false, false);
        aN = __builtin_amdgcn_wmma_f32_16x16x32_bf16(false, fa.v, false, fn.v, (short)0, aN, false, false);
      }
      // C layout: VGPR e -> row m0+e (lanes 0-15) / m0+e+8 (lanes 16-31), col = lane%16.
#pragma unroll
      for (int e = 0; e < 8; ++e) {
        const int row  = m0 + e + (hi << 3);
        const float hp = hreg[mt * 8 + e];
        const float xv = x_lds[row];
        const float rr = sigm_fast(fmaf(xv, wih_l[c],      brz_l[c])      + aR[e]);
        const float zz = sigm_fast(fmaf(xv, wih_l[16 + c], brz_l[16 + c]) + aZ[e]);
        const float nn = tanh_fast(fmaf(xv, wih_l[32 + c], bin_l[c]) + rr * (aN[e] + bhn_l[c]));
        float hn = fmaf(zz, hp - nn, nn);          // (1-z)*n + z*h
        if (enc && t >= slreg[mt * 8 + e]) hn = hp;   // packed-sequence masking
        hreg[mt * 8 + e] = hn;
        hnb[(size_t)row * Hn + j] = (unsigned short)f2bf(hn);
        if (!enc) {
          // Partial of h_new . Wout over this j-block (fixed order -> deterministic).
          float p = hn * wout_l[c];
          p += __shfl_xor(p, 1);
          p += __shfl_xor(p, 2);
          p += __shfl_xor(p, 4);
          p += __shfl_xor(p, 8);
          if (c == 0) xp_out[wg * Bn + row] = p;
        }
      }
    }
  };

  // ---- Init: zero bf16 h0, copy input to d_out, stage encoder weights
  for (int i = gtid; i < Bn * Hn; i += GSZ) hb0[i] = 0;
  for (int i = gtid; i < Bn * Tn; i += GSZ) out_in[i] = input[i];
  stage(eWhh, eWih, ebih, ebhh);
  gsync();

  unsigned short* hb[2] = {hb0, hb1};
  int cur = 0;

  // ---- Encoder: 1024 masked GRU steps
  for (int t = 0; t < Tn; ++t) {
    x_lds[tid] = input[(size_t)tid * Tn + t];
    __syncthreads();
    do_step(hb[cur], hb[cur ^ 1], nullptr, true, t);
    gsync();
    cur ^= 1;
  }

  // features = h_final ([1,B,H] row-major), straight from registers
  {
    const int c = lane & 15, hi = lane >> 4;
#pragma unroll
    for (int mt = 0; mt < 2; ++mt)
#pragma unroll
      for (int e = 0; e < 8; ++e) {
        const int row = ((wave * 2 + mt) << 4) + e + (hi << 3);
        out_feat[(size_t)row * Hn + j0 + c] = hreg[mt * 8 + e];
      }
  }

  // ---- Decoder weight staging
  stage(dWhh, dWih, dbih, dbhh);
  if (tid < 16) wout_l[tid] = Wout[j0 + tid];
  const float boutv = bout[0];
  __syncthreads();

  // ---- Decoder: feed y back as next x; y_t written as output[:, t]
  for (int t = 0; t < Tn; ++t) {
    {
      float xv = 0.0f;
      if (t > 0) {
        const float* xp = xpart + (size_t)((t - 1) & 1) * NWG * Bn;
        float s = boutv;
        for (int jb = 0; jb < NWG; ++jb) s += xp[jb * Bn + tid];
        xv = s;
        if (wg == 0) out_y[(size_t)tid * Tn + (t - 1)] = s;
      }
      x_lds[tid] = xv;
    }
    __syncthreads();
    do_step(hb[cur], hb[cur ^ 1], xpart + (size_t)(t & 1) * NWG * Bn, false, t);
    gsync();
    cur ^= 1;
  }
  if (wg == 0) {
    const float* xp = xpart + (size_t)((Tn - 1) & 1) * NWG * Bn;
    float s = boutv;
    for (int jb = 0; jb < NWG; ++jb) s += xp[jb * Bn + tid];
    out_y[(size_t)tid * Tn + (Tn - 1)] = s;
  }
  gsync();

  // ---- Masked MSE loss (fixed-order two-level reduction -> deterministic)
  float se = 0.0f, cn = 0.0f;
  for (int i = gtid; i < Bn * Tn; i += GSZ) {
    const int b = i >> 10;            // / Tn
    const int tt = i & (Tn - 1);      // % Tn
    if (tt < seqlen[b]) {
      const float d = input[i] - out_y[i];
      se += d * d;
      cn += 1.0f;
    }
  }
  red1[tid] = se; red2[tid] = cn;
  __syncthreads();
  for (int s = BLK / 2; s > 0; s >>= 1) {
    if (tid < s) { red1[tid] += red1[tid + s]; red2[tid] += red2[tid + s]; }
    __syncthreads();
  }
  if (tid == 0) { lpart[wg] = red1[0]; lpart[NWG + wg] = red2[0]; }
  gsync();
  if (wg == 0 && tid == 0) {
    float S = 0.0f, C = 0.0f;
    for (int k = 0; k < NWG; ++k) { S += lpart[k]; C += lpart[NWG + k]; }
    out_loss[0] = S / C;
  }
}

extern "C" void kernel_launch(void* const* d_in, const int* in_sizes, int n_in,
                              void* d_out, int out_size, void* d_ws, size_t ws_size,
                              hipStream_t stream) {
  (void)in_sizes; (void)n_in; (void)out_size; (void)ws_size;
  const float* input = (const float*)d_in[0];
  const int*   seql  = (const int*)d_in[1];
  const float* eWih  = (const float*)d_in[2];
  const float* eWhh  = (const float*)d_in[3];
  const float* ebih  = (const float*)d_in[4];
  const float* ebhh  = (const float*)d_in[5];
  const float* dWih  = (const float*)d_in[6];
  const float* dWhh  = (const float*)d_in[7];
  const float* dbih  = (const float*)d_in[8];
  const float* dbhh  = (const float*)d_in[9];
  const float* Wout  = (const float*)d_in[10];
  const float* bout  = (const float*)d_in[11];

  // d_out: [loss(1) | features(B*H) | input(B*T) | output(B*T)]
  float* out      = (float*)d_out;
  float* out_loss = out;
  float* out_feat = out + 1;
  float* out_in   = out + 1 + Bn * Hn;
  float* out_y    = out + 1 + Bn * Hn + Bn * Tn;

  // Workspace carve-up (~0.6 MB)
  char* p = (char*)d_ws;
  unsigned* syncc = (unsigned*)p;                 p += 256;   // zeroed per launch
  float* lpart    = (float*)p;                    p += 256;
  float* xpart    = (float*)p;                    p += (size_t)2 * NWG * Bn * sizeof(float);
  unsigned short* hb0 = (unsigned short*)p;       p += (size_t)Bn * Hn * sizeof(unsigned short);
  unsigned short* hb1 = (unsigned short*)p;       p += (size_t)Bn * Hn * sizeof(unsigned short);

  // Graph-capture-legal reset of the grid-sync counter.
  hipMemsetAsync(syncc, 0, 256, stream);

  gru_autoencoder_kernel<<<dim3(NWG), dim3(BLK), 0, stream>>>(
      input, seql, eWih, eWhh, ebih, ebhh, dWih, dWhh, dbih, dbhh, Wout, bout,
      out_loss, out_feat, out_in, out_y, hb0, hb1, xpart, lpart, syncc);
}
